// AttentionBlock_25847113187450
// MI455X (gfx1250) — matrix operand order repaired and checked by reading, not compiled, not run on hardware
//
#include <hip/hip_runtime.h>
#include <hip/hip_bf16.h>

// MI455X / gfx1250 attention block, bf16 WMMA (f32 accumulate) throughout.
// Pipeline:
//   K0: LDS-tiled transpose+downcast  x -> Xbf (B,N,C); W_qkv -> W_qkv^T; W_out -> W_out^T
//   K1: QKV GEMM (32x64 per wave, B-fragment reuse x2) -> Q*scale, K, V^T
//   K2: flash attention per (b,h): S = Q K^T (pre-scaled), online softmax, O = P V
//   K3: out GEMM (32x64 per wave) + b_out + residual + (B,N,C)->(B,C,N) transpose

typedef __bf16 bf16;
typedef __attribute__((ext_vector_type(16))) __bf16 v16bf;
typedef __attribute__((ext_vector_type(8)))  __bf16 v8bf;
typedef __attribute__((ext_vector_type(8)))  float  v8f;
typedef __attribute__((ext_vector_type(4)))  float  v4f;

// ---------------------------------------------------------------------------
// Fragment loaders for wave32 WMMA 16x16x32 bf16 (layouts per CDNA5 ISA 7.12.2)
// A (16x32, MxK), row-major source:
//   lane L: m = mtile + (L&15), half = L>>4
//   a[0..7]  = row m, k = koff + 8*half .. +7        (contiguous 16B)
//   a[8..15] = row m, k = koff + 16 + 8*half .. +7   (contiguous 16B)
__device__ __forceinline__ v16bf load_frag_a(const bf16* __restrict__ base, int ld,
                                             int mtile, int koff, int lane) {
  const int m = mtile + (lane & 15);
  const int half = lane >> 4;
  const bf16* p = base + (size_t)m * ld + koff;
  union { v16bf v; v8bf h[2]; } u;
  u.h[0] = *(const v8bf*)(p + 8 * half);
  u.h[1] = *(const v8bf*)(p + 16 + 8 * half);
  return u.v;
}

// B (32x16, KxN) from *transposed* storage: element B[k][n] lives at base[n*ld + k].
//   lane L: n = ntile + (L&15), half = L>>4
//   b[0..7] = k = koff + 8*half .. +7, b[8..15] = k = koff + 16 + 8*half .. +7   (two b128 loads)
__device__ __forceinline__ v16bf load_frag_bT(const bf16* __restrict__ base, int ld,
                                              int ntile, int koff, int lane) {
  const int n = ntile + (lane & 15);
  const int half = lane >> 4;
  const bf16* p = base + (size_t)n * ld + koff;
  union { v16bf v; v8bf h[2]; } u;
  u.h[0] = *(const v8bf*)(p + 8 * half);
  u.h[1] = *(const v8bf*)(p + 16 + 8 * half);
  return u.v;
}

__device__ __forceinline__ v8f wmma_bf16(v16bf a, v16bf b, v8f c) {
  return __builtin_amdgcn_wmma_f32_16x16x32_bf16(false, a, false, b, (short)0, c,
                                                 false, false);
}

// ---------------------------------------------------------------------------
// K0: batched LDS-tiled transpose + f32->bf16.
// in: (batch, R, C) f32 row-major -> out: (batch, C, R) bf16 row-major.
// R, C multiples of 32. Block (32,8) handles one 32x32 tile; coalesced both ways.
__global__ __launch_bounds__(256) void transpose_to_bf16(
    const float* __restrict__ in, bf16* __restrict__ out, int R, int C) {
  __shared__ float tile[32][33];
  const int tx = threadIdx.x, ty = threadIdx.y;
  const int c0 = blockIdx.x * 32;  // input col tile
  const int r0 = blockIdx.y * 32;  // input row tile
  const size_t base = (size_t)blockIdx.z * R * C;
#pragma unroll
  for (int i = 0; i < 32; i += 8)
    tile[ty + i][tx] = in[base + (size_t)(r0 + ty + i) * C + c0 + tx];
  __syncthreads();
#pragma unroll
  for (int i = 0; i < 32; i += 8)
    out[base + (size_t)(c0 + ty + i) * R + r0 + tx] = (bf16)tile[tx][ty + i];
}

// ---------------------------------------------------------------------------
// K1: QKV GEMM.  M = B*N = 16384, Ncols = 1536, K = 512.
// 8 waves/block arranged 2(M) x 4(N); each wave computes a 32x64 tile:
// two A fragments share four B fragments -> 0.75 fragment loads per WMMA.
__global__ __launch_bounds__(256) void qkv_gemm(
    const bf16* __restrict__ Xbf,       // (16384, 512)
    const bf16* __restrict__ Wt,        // (1536, 512) = W_qkv^T
    const float* __restrict__ bias,     // (1536)
    bf16* __restrict__ Qb,              // (16,8,1024,64)  pre-scaled by 1/8
    bf16* __restrict__ Kb,              // (16,8,1024,64)
    bf16* __restrict__ Vt) {            // (16,8,64,1024)
  const int lane = threadIdx.x & 31;
  const int wv = threadIdx.x >> 5;
  const int wm = wv & 1, wn = wv >> 1;
  const int mtile = blockIdx.x * 64 + wm * 32;
  const int ntile = blockIdx.y * 256 + wn * 64;
  const int half = lane >> 4, l16 = lane & 15;

  v8f acc[2][4];
#pragma unroll
  for (int mi = 0; mi < 2; ++mi)
#pragma unroll
    for (int c = 0; c < 4; ++c)
#pragma unroll
      for (int r = 0; r < 8; ++r) acc[mi][c][r] = 0.0f;

#pragma unroll
  for (int kk = 0; kk < 512; kk += 32) {
    __builtin_prefetch(Xbf + (size_t)(mtile + l16) * 512 + kk + 32, 0, 1);
    const v16bf a0 = load_frag_a(Xbf, 512, mtile, kk, lane);
    const v16bf a1 = load_frag_a(Xbf, 512, mtile + 16, kk, lane);
#pragma unroll
    for (int c = 0; c < 4; ++c) {
      const v16bf bfr = load_frag_bT(Wt, 512, ntile + 16 * c, kk, lane);
      acc[0][c] = wmma_bf16(a0, bfr, acc[0][c]);
      acc[1][c] = wmma_bf16(a1, bfr, acc[1][c]);
    }
  }

  // Scatter into Q / K / V^T.  Column j -> head h = j/192, which = (j%192)/64.
#pragma unroll
  for (int mi = 0; mi < 2; ++mi) {
    const int row0 = mtile + mi * 16 + 8 * half;
    const int bb = row0 >> 10;
    const int nn0 = row0 & 1023;
#pragma unroll
    for (int c4 = 0; c4 < 4; ++c4) {
      const int j = ntile + c4 * 16 + l16;
      const int hh = j / 192;
      const int ww = j - hh * 192;
      const int which = ww >> 6;
      const int d = ww & 63;
      const float bj = bias[j];
      if (which == 2) {
        // V^T rows: 8 consecutive token columns per lane -> one 16B store
        v8bf pack;
#pragma unroll
        for (int r = 0; r < 8; ++r) pack[r] = (bf16)(acc[mi][c4][r] + bj);
        *(v8bf*)&Vt[((size_t)(bb * 8 + hh) * 64 + d) * 1024 + nn0] = pack;
      } else if (which == 0) {
#pragma unroll
        for (int r = 0; r < 8; ++r)
          Qb[((size_t)(bb * 8 + hh) * 1024 + nn0 + r) * 64 + d] =
              (bf16)((acc[mi][c4][r] + bj) * 0.125f);
      } else {
#pragma unroll
        for (int r = 0; r < 8; ++r)
          Kb[((size_t)(bb * 8 + hh) * 1024 + nn0 + r) * 64 + d] =
              (bf16)(acc[mi][c4][r] + bj);
      }
    }
  }
}

// ---------------------------------------------------------------------------
// K2: flash attention.  grid = (16 qtiles, 8 heads, 16 batch), 4 waves/block,
// each wave owns 16 query rows; loop over 1024 keys in tiles of 32.
// No asm fence: same-wave DS ops are hardware in-order and Pbuf aliasing keeps
// the compiler from reordering the LDS bounce, while provably-disjoint global
// loads of the next tile (unroll 2) hoist above it to hide latency.
__global__ __launch_bounds__(128) void attn_kernel(
    const bf16* __restrict__ Qb, const bf16* __restrict__ Kb,
    const bf16* __restrict__ Vt, bf16* __restrict__ Att) {
  const int lane = threadIdx.x & 31;
  const int w = threadIdx.x >> 5;
  const int qt = blockIdx.x;
  const int h = blockIdx.y;
  const int b = blockIdx.z;
  const int half = lane >> 4, l16 = lane & 15;

  const size_t bh = (size_t)(b * 8 + h);
  const bf16* Qbh = Qb + bh * 1024 * 64;
  const bf16* Kbh = Kb + bh * 1024 * 64;
  const bf16* Vbh = Vt + bh * 64 * 1024;

  const int qrow = qt * 64 + w * 16;

  const v16bf aq0 = load_frag_a(Qbh, 64, qrow, 0, lane);
  const v16bf aq1 = load_frag_a(Qbh, 64, qrow, 32, lane);

  v8f o0, o1, o2, o3;
  float mrow[8], srow[8];
#pragma unroll
  for (int r = 0; r < 8; ++r) {
    o0[r] = 0.f; o1[r] = 0.f; o2[r] = 0.f; o3[r] = 0.f;
    mrow[r] = -1e30f;
    srow[r] = 0.f;
  }

  // Per-wave P bounce buffer: 16 rows x 32 cols, pitch 40 keeps rows 16B-aligned.
  __shared__ __align__(16) bf16 Pbuf[4][16][40];

#pragma unroll 2
  for (int kt = 0; kt < 1024; kt += 32) {
    __builtin_prefetch(Kbh + (size_t)(kt + 32 + l16) * 64, 0, 1);
    // ---- issue all fragment loads for this tile up-front ----
    v16bf bk00 = load_frag_bT(Kbh, 64, kt, 0, lane);
    v16bf bk01 = load_frag_bT(Kbh, 64, kt, 32, lane);
    v16bf bk10 = load_frag_bT(Kbh, 64, kt + 16, 0, lane);
    v16bf bk11 = load_frag_bT(Kbh, 64, kt + 16, 32, lane);
    v16bf bv0 = load_frag_bT(Vbh, 1024, 0,  kt, lane);
    v16bf bv1 = load_frag_bT(Vbh, 1024, 16, kt, lane);
    v16bf bv2 = load_frag_bT(Vbh, 1024, 32, kt, lane);
    v16bf bv3 = load_frag_bT(Vbh, 1024, 48, kt, lane);

    // ---- S tiles: 16 queries x 32 keys ----
    v8f s0, s1;
#pragma unroll
    for (int r = 0; r < 8; ++r) { s0[r] = 0.f; s1[r] = 0.f; }
    s0 = wmma_bf16(aq0, bk00, s0);
    s0 = wmma_bf16(aq1, bk01, s0);
    s1 = wmma_bf16(aq0, bk10, s1);
    s1 = wmma_bf16(aq1, bk11, s1);

    // ---- online softmax (row stats across 16 lanes of each half) ----
    float corr[8];
#pragma unroll
    for (int r = 0; r < 8; ++r) {
      float t = fmaxf(s0[r], s1[r]);
      t = fmaxf(t, __shfl_xor(t, 1, 32));
      t = fmaxf(t, __shfl_xor(t, 2, 32));
      t = fmaxf(t, __shfl_xor(t, 4, 32));
      t = fmaxf(t, __shfl_xor(t, 8, 32));
      const float mnew = fmaxf(mrow[r], t);
      corr[r] = __expf(mrow[r] - mnew);
      mrow[r] = mnew;
      const float p0 = __expf(s0[r] - mnew);
      const float p1 = __expf(s1[r] - mnew);
      s0[r] = p0;
      s1[r] = p1;
      float ps = p0 + p1;
      ps += __shfl_xor(ps, 1, 32);
      ps += __shfl_xor(ps, 2, 32);
      ps += __shfl_xor(ps, 4, 32);
      ps += __shfl_xor(ps, 8, 32);
      srow[r] = srow[r] * corr[r] + ps;
    }

    // rescale O, spill P (C layout) to LDS
#pragma unroll
    for (int r = 0; r < 8; ++r) {
      o0[r] *= corr[r]; o1[r] *= corr[r]; o2[r] *= corr[r]; o3[r] *= corr[r];
      Pbuf[w][8 * half + r][l16]      = (bf16)s0[r];
      Pbuf[w][8 * half + r][l16 + 16] = (bf16)s1[r];
    }

    // re-read P as A fragment (16x32); same-wave DS ops are in-order.
    const bf16* pp = &Pbuf[w][l16][0];
    union { v16bf v; v8bf hh[2]; } up;
    up.hh[0] = *(const v8bf*)(pp + 8 * half);
    up.hh[1] = *(const v8bf*)(pp + 16 + 8 * half);
    const v16bf ap = up.v;

    // ---- O += P @ V ----
    o0 = wmma_bf16(ap, bv0, o0);
    o1 = wmma_bf16(ap, bv1, o1);
    o2 = wmma_bf16(ap, bv2, o2);
    o3 = wmma_bf16(ap, bv3, o3);
  }

  // normalize + store Att (B, N, H*D) bf16
#pragma unroll
  for (int r = 0; r < 8; ++r) {
    const float inv = 1.0f / srow[r];
    const size_t rowbase =
        ((size_t)(b * 1024 + qrow + 8 * half + r)) * 512 + h * 64;
    Att[rowbase +  0 + l16] = (bf16)(o0[r] * inv);
    Att[rowbase + 16 + l16] = (bf16)(o1[r] * inv);
    Att[rowbase + 32 + l16] = (bf16)(o2[r] * inv);
    Att[rowbase + 48 + l16] = (bf16)(o3[r] * inv);
  }
}

// ---------------------------------------------------------------------------
// K3: out projection.  M = 16384, Ncols = 512, K = 512.
// Same 32x64-per-wave tiling as K1.  Epilogue fuses + b_out + residual x and
// the (B,N,C)->(B,C,N) transpose with b128 residual loads / output stores.
__global__ __launch_bounds__(256) void out_gemm(
    const bf16* __restrict__ Att,       // (16384, 512)
    const bf16* __restrict__ Wot,       // (512, 512) = W_out^T
    const float* __restrict__ bias,     // (512)
    const float* __restrict__ x,        // (16, 512, 1024)
    float* __restrict__ out) {          // (16, 512, 1024)
  const int lane = threadIdx.x & 31;
  const int wv = threadIdx.x >> 5;
  const int wm = wv & 1, wn = wv >> 1;
  const int mtile = blockIdx.x * 64 + wm * 32;
  const int ntile = blockIdx.y * 256 + wn * 64;
  const int half = lane >> 4, l16 = lane & 15;

  v8f acc[2][4];
#pragma unroll
  for (int mi = 0; mi < 2; ++mi)
#pragma unroll
    for (int c = 0; c < 4; ++c)
#pragma unroll
      for (int r = 0; r < 8; ++r) acc[mi][c][r] = 0.0f;

#pragma unroll
  for (int kk = 0; kk < 512; kk += 32) {
    __builtin_prefetch(Att + (size_t)(mtile + l16) * 512 + kk + 32, 0, 1);
    const v16bf a0 = load_frag_a(Att, 512, mtile, kk, lane);
    const v16bf a1 = load_frag_a(Att, 512, mtile + 16, kk, lane);
#pragma unroll
    for (int c = 0; c < 4; ++c) {
      const v16bf bfr = load_frag_bT(Wot, 512, ntile + 16 * c, kk, lane);
      acc[0][c] = wmma_bf16(a0, bfr, acc[0][c]);
      acc[1][c] = wmma_bf16(a1, bfr, acc[1][c]);
    }
  }

#pragma unroll
  for (int mi = 0; mi < 2; ++mi) {
    const int row0 = mtile + mi * 16 + 8 * half;
    const int bb = row0 >> 10;
    const int nn0 = row0 & 1023;
#pragma unroll
    for (int c4 = 0; c4 < 4; ++c4) {
      const int cc = ntile + c4 * 16 + l16;
      const float bj = bias[cc];
      const size_t oidx = ((size_t)bb * 512 + cc) * 1024 + nn0;
      const v4f x0 = *(const v4f*)&x[oidx];
      const v4f x1 = *(const v4f*)&x[oidx + 4];
      v4f y0, y1;
#pragma unroll
      for (int r = 0; r < 4; ++r) {
        y0[r] = acc[mi][c4][r] + bj + x0[r];
        y1[r] = acc[mi][c4][4 + r] + bj + x1[r];
      }
      *(v4f*)&out[oidx] = y0;
      *(v4f*)&out[oidx + 4] = y1;
    }
  }
}

// ---------------------------------------------------------------------------
extern "C" void kernel_launch(void* const* d_in, const int* in_sizes, int n_in,
                              void* d_out, int out_size, void* d_ws, size_t ws_size,
                              hipStream_t stream) {
  (void)in_sizes; (void)n_in; (void)out_size; (void)ws_size;
  const float* x     = (const float*)d_in[0];  // (16, 512, 32, 32)
  const float* W_qkv = (const float*)d_in[1];  // (512, 1536)
  const float* b_qkv = (const float*)d_in[2];  // (1536)
  const float* W_out = (const float*)d_in[3];  // (512, 512)
  const float* b_out = (const float*)d_in[4];  // (512)
  float* out = (float*)d_out;                  // (16, 512, 32, 32)

  char* ws = (char*)d_ws;
  const size_t SZ_XBF   = (size_t)16 * 1024 * 512 * 2;
  const size_t SZ_WQKVT = (size_t)1536 * 512 * 2;
  const size_t SZ_WOUTT = (size_t)512 * 512 * 2;
  const size_t SZ_HEAD  = (size_t)16 * 8 * 1024 * 64 * 2;
  const size_t SZ_ATT   = (size_t)16 * 1024 * 512 * 2;

  size_t off = 0;
  bf16* Xbf   = (bf16*)(ws + off); off += SZ_XBF;
  bf16* Wqkvt = (bf16*)(ws + off); off += SZ_WQKVT;
  bf16* Woutt = (bf16*)(ws + off); off += SZ_WOUTT;
  bf16* Qb    = (bf16*)(ws + off); off += SZ_HEAD;
  bf16* Kb    = (bf16*)(ws + off); off += SZ_HEAD;
  bf16* Vt    = (bf16*)(ws + off); off += SZ_HEAD;
  bf16* Att   = (bf16*)(ws + off); off += SZ_ATT;

  // K0: layout prep (coalesced LDS-tiled transpose + bf16 downcast)
  transpose_to_bf16<<<dim3(32, 16, 16), dim3(32, 8), 0, stream>>>(x, Xbf, 512, 1024);
  transpose_to_bf16<<<dim3(48, 16, 1), dim3(32, 8), 0, stream>>>(W_qkv, Wqkvt, 512, 1536);
  transpose_to_bf16<<<dim3(16, 16, 1), dim3(32, 8), 0, stream>>>(W_out, Woutt, 512, 512);

  // K1: QKV projection -> Q (pre-scaled), K, V^T   (block tile 64M x 256N)
  qkv_gemm<<<dim3(256, 6), 256, 0, stream>>>(Xbf, Wqkvt, b_qkv, Qb, Kb, Vt);

  // K2: flash attention
  attn_kernel<<<dim3(16, 8, 16), 128, 0, stream>>>(Qb, Kb, Vt, Att);

  // K3: out projection + bias + residual + final transpose
  out_gemm<<<dim3(256, 2), 256, 0, stream>>>(Att, Woutt, b_out, x, out);
}
